// AttentionLayer_86715389706633
// MI455X (gfx1250) — matrix operand order repaired and checked by reading, not compiled, not run on hardware
//
#include <hip/hip_runtime.h>

// ---------------------------------------------------------------------------
// Problem constants (from the reference):  B=4, SQ=SK=2048, D=UNITS=1024
// ---------------------------------------------------------------------------
#define B_   4
#define SQ_  2048
#define SK_  2048
#define D_   1024
#define U_   1024

typedef __attribute__((ext_vector_type(16))) __bf16 v16bf;
typedef __attribute__((ext_vector_type(8)))  __bf16 v8bf;
typedef __attribute__((ext_vector_type(8)))  float  v8f;
typedef __attribute__((ext_vector_type(4)))  float  v4f;

static __device__ __forceinline__ v16bf cat8(v8bf lo, v8bf hi) {
  return __builtin_shufflevector(lo, hi, 0,1,2,3,4,5,6,7,8,9,10,11,12,13,14,15);
}

static __device__ __forceinline__ v8f wmma_bf16(v16bf a, v16bf b, v8f c) {
  return __builtin_amdgcn_wmma_f32_16x16x32_bf16(false, a, false, b,
                                                 (short)0, c, false, false);
}

// A fragment 16x32 (MxK) bf16 from a row-major [.][ld] source, base at [m0][k0].
// Lane<16: row=lane, K runs {0..7, 16..23}; lane>=16: row=lane-16, K runs {8..15, 24..31}.
static __device__ __forceinline__ v16bf load_a_bf16(const __bf16* base, int ld) {
  int lane = threadIdx.x & 31;
  int half = lane >> 4;
  int row  = lane & 15;
  const __bf16* p = base + (size_t)row * ld + half * 8;
  v8bf lo = *(const v8bf*)(p);
  v8bf hi = *(const v8bf*)(p + 16);
  return cat8(lo, hi);
}

// Same A fragment but from an fp32 source, converting to bf16 in flight.
static __device__ __forceinline__ v16bf load_a_f32(const float* base, int ld) {
  int lane = threadIdx.x & 31;
  int half = lane >> 4;
  int row  = lane & 15;
  const float* p = base + (size_t)row * ld + half * 8;
  v4f a0 = *(const v4f*)(p + 0);
  v4f a1 = *(const v4f*)(p + 4);
  v4f b0 = *(const v4f*)(p + 16);
  v4f b1 = *(const v4f*)(p + 20);
  v16bf r;
#pragma unroll
  for (int i = 0; i < 4; ++i) {
    r[i]      = (__bf16)a0[i];
    r[4 + i]  = (__bf16)a1[i];
    r[8 + i]  = (__bf16)b0[i];
    r[12 + i] = (__bf16)b1[i];
  }
  return r;
}

// B fragment 32x16 (KxN) bf16 sourced from a TRANSPOSED row-major [N][K]
// buffer, base at [n0][k0].  Lane<16: col=lane, K runs {0..7, 16..23};
// lane>=16: col=lane-16, K runs {8..15, 24..31}.
static __device__ __forceinline__ v16bf load_b_bf16(const __bf16* baseT, int ld) {
  int lane = threadIdx.x & 31;
  int half = lane >> 4;
  int col  = lane & 15;
  const __bf16* p = baseT + (size_t)col * ld + half * 8;
  v8bf lo = *(const v8bf*)(p);
  v8bf hi = *(const v8bf*)(p + 16);
  return cat8(lo, hi);
}

// ---------------------------------------------------------------------------
// Prep: W [D][U] fp32 -> Wt [U][D] bf16  (B-fragment friendly layout)
// ---------------------------------------------------------------------------
__global__ void k_transpose_w(const float* __restrict__ W, __bf16* __restrict__ Wt) {
  int tid = blockIdx.x * blockDim.x + threadIdx.x;   // over U_*D_, write-coalesced
  int u = tid / D_;
  int d = tid - u * D_;
  Wt[(size_t)u * D_ + d] = (__bf16)W[(size_t)d * U_ + u];
}

// Prep: V [B][SK][D] fp32 -> Vt [B][D][SK] bf16
__global__ void k_transpose_v(const float* __restrict__ V, __bf16* __restrict__ Vt) {
  int tid = blockIdx.x * blockDim.x + threadIdx.x;   // over B_*D_*SK_, write-coalesced
  int b = tid / (D_ * SK_);
  int r = tid - b * (D_ * SK_);
  int d = r / SK_;
  int s = r - d * SK_;
  Vt[tid] = (__bf16)V[((size_t)b * SK_ + s) * D_ + d];
}

// ---------------------------------------------------------------------------
// Projection GEMM:  out_bf16[M][U] = A_f32[M][D] @ Wt_bf16[U][D]^T + bias
// One wave computes a 32x64 tile (8 accumulators), K-loop step 32,
// software-pipelined (next fragments load while current WMMAs execute).
// All 8 waves of a block share the same m-tile -> A rows hit WGP$/L2.
// ---------------------------------------------------------------------------
__global__ __launch_bounds__(256) void k_proj(const float* __restrict__ A,
                                              const __bf16* __restrict__ Wt,
                                              const float* __restrict__ bias,
                                              __bf16* __restrict__ out) {
  int w     = blockIdx.x * (blockDim.x >> 5) + (threadIdx.x >> 5);
  int mtile = w >> 4;                 // 256 m-tiles of 32 (M = B*SQ = 8192)
  int ntile = w & 15;                 // 16 n-tiles of 64 (N = 1024)
  int m0 = mtile * 32, n0 = ntile * 64;
  int lane = threadIdx.x & 31;

  const float*  a0p = A + (size_t)m0 * D_;
  const float*  a1p = A + (size_t)(m0 + 16) * D_;
  const __bf16* bp  = Wt + (size_t)n0 * D_;

  v8f acc[2][4] = {};

  // Prologue: load k-step 0 fragments.
  v16bf a0 = load_a_f32(a0p, D_);
  v16bf a1 = load_a_f32(a1p, D_);
  v16bf b[4];
#pragma unroll
  for (int j = 0; j < 4; ++j) b[j] = load_b_bf16(bp + (size_t)j * 16 * D_, D_);

  for (int kk = 0; kk < D_ - 32; kk += 32) {
    int nk = kk + 32;
    // Prefetch one further step ahead on the A stream (global_prefetch_b8).
    __builtin_prefetch(a0p + (size_t)(lane & 15) * D_ + nk + 32, 0, 3);
    // Load next-step fragments into fresh registers (overlaps with WMMAs).
    v16bf na0 = load_a_f32(a0p + nk, D_);
    v16bf na1 = load_a_f32(a1p + nk, D_);
    v16bf nb[4];
#pragma unroll
    for (int j = 0; j < 4; ++j) nb[j] = load_b_bf16(bp + (size_t)j * 16 * D_ + nk, D_);
#pragma unroll
    for (int j = 0; j < 4; ++j) {
      acc[0][j] = wmma_bf16(a0, b[j], acc[0][j]);
      acc[1][j] = wmma_bf16(a1, b[j], acc[1][j]);
    }
    a0 = na0; a1 = na1;
#pragma unroll
    for (int j = 0; j < 4; ++j) b[j] = nb[j];
  }
#pragma unroll
  for (int j = 0; j < 4; ++j) {
    acc[0][j] = wmma_bf16(a0, b[j], acc[0][j]);
    acc[1][j] = wmma_bf16(a1, b[j], acc[1][j]);
  }

  int rofs = (lane < 16) ? 0 : 8;
  int c    = lane & 15;
#pragma unroll
  for (int i = 0; i < 2; ++i)
#pragma unroll
    for (int j = 0; j < 4; ++j) {
      int   col = n0 + j * 16 + c;
      float bv  = bias[col];
#pragma unroll
      for (int r = 0; r < 8; ++r)
        out[(size_t)(m0 + i * 16 + r + rofs) * U_ + col] = (__bf16)(acc[i][j][r] + bv);
    }
}

// ---------------------------------------------------------------------------
// Score + softmax:  one workgroup per (batch, 16 q-rows).
// LDS: l1 tile (16x1024 bf16, 32KB) + score strip (16x2048 f32, 128KB) + 1KB.
// Each wave processes groups of 4 k-tiles (4 accumulators sharing one LDS
// A-fragment); global B stream double-buffered.
// ---------------------------------------------------------------------------
__global__ __launch_bounds__(256) void k_score_softmax(const __bf16* __restrict__ L1,
                                                       const __bf16* __restrict__ L2,
                                                       __bf16* __restrict__ P) {
  extern __shared__ char smem[];
  __bf16* l1t = (__bf16*)smem;                                   // 16*U_ bf16
  float*  sc  = (float*)(smem + 16 * U_ * 2);                    // 16*SK_ f32
  float*  red = (float*)(smem + 16 * U_ * 2 + 16 * SK_ * 4);     // 16*16 f32

  int b   = blockIdx.x >> 7;          // SQ_/16 = 128 q-tiles per batch
  int qt  = blockIdx.x & 127;
  int q0  = qt * 16;
  int tid = threadIdx.x;

  // Stage the 16x1024 l1 tile into LDS (b128 copies).
  const v8bf* src = (const v8bf*)(L1 + ((size_t)b * SQ_ + q0) * U_);
  v8bf*       dst = (v8bf*)l1t;
  for (int i = tid; i < 16 * U_ / 8; i += blockDim.x) dst[i] = src[i];
  __syncthreads();

  const float inv_scale = 0.022097086912079608f;   // 1/sqrt(SK=2048)
  int wave = tid >> 5;
  int lane = tid & 31;

  // 32 groups of 4 k-tiles (64 cols); each of 8 waves takes 4 groups.
  for (int g = wave; g < SK_ / 64; g += 8) {
    int k0 = g * 64;
    const __bf16* bbase = L2 + ((size_t)b * SK_ + k0) * U_;   // [SK][U] == Bt[N][K]

    v8f acc[4] = {};
    v16bf a = load_a_bf16(l1t, U_);
    v16bf bfr[4];
#pragma unroll
    for (int j = 0; j < 4; ++j) bfr[j] = load_b_bf16(bbase + (size_t)j * 16 * U_, U_);

    for (int u = 0; u < U_ - 32; u += 32) {
      int nu = u + 32;
      v16bf na = load_a_bf16(l1t + nu, U_);
      v16bf nb[4];
#pragma unroll
      for (int j = 0; j < 4; ++j) nb[j] = load_b_bf16(bbase + (size_t)j * 16 * U_ + nu, U_);
#pragma unroll
      for (int j = 0; j < 4; ++j) acc[j] = wmma_bf16(a, bfr[j], acc[j]);
      a = na;
#pragma unroll
      for (int j = 0; j < 4; ++j) bfr[j] = nb[j];
    }
#pragma unroll
    for (int j = 0; j < 4; ++j) acc[j] = wmma_bf16(a, bfr[j], acc[j]);

    int rofs = (lane < 16) ? 0 : 8;
#pragma unroll
    for (int j = 0; j < 4; ++j) {
      int c = k0 + j * 16 + (lane & 15);
#pragma unroll
      for (int r = 0; r < 8; ++r)
        sc[(size_t)(r + rofs) * SK_ + c] = acc[j][r] * inv_scale;
    }
  }
  __syncthreads();

  // Row softmax over 2048 cols: 16 threads per row, 128 cols per thread.
  int row = tid >> 4;
  int seg = tid & 15;
  float* rp = sc + (size_t)row * SK_ + seg * (SK_ / 16);

  float m = -3.4e38f;
  for (int i = 0; i < SK_ / 16; ++i) m = fmaxf(m, rp[i]);
  red[row * 16 + seg] = m;
  __syncthreads();
  m = -3.4e38f;
  for (int i = 0; i < 16; ++i) m = fmaxf(m, red[row * 16 + i]);
  __syncthreads();

  float s = 0.f;
  for (int i = 0; i < SK_ / 16; ++i) {
    float e = __expf(rp[i] - m);
    rp[i] = e;
    s += e;
  }
  red[row * 16 + seg] = s;
  __syncthreads();
  s = 0.f;
  for (int i = 0; i < 16; ++i) s += red[row * 16 + i];
  float inv = 1.0f / s;

  __bf16* op = P + ((size_t)b * SQ_ + q0 + row) * SK_ + seg * (SK_ / 16);
  for (int i = 0; i < SK_ / 16; ++i) op[i] = (__bf16)(rp[i] * inv);
}

// ---------------------------------------------------------------------------
// att GEMM:  out_f32[B*SQ][D] = P_bf16[B*SQ][SK] @ Vt_bf16[b][D][SK]^T
// 32x64 tile per wave, software-pipelined like k_proj.
// ---------------------------------------------------------------------------
__global__ __launch_bounds__(256) void k_att(const __bf16* __restrict__ P,
                                             const __bf16* __restrict__ Vt,
                                             float* __restrict__ out) {
  int w     = blockIdx.x * (blockDim.x >> 5) + (threadIdx.x >> 5);
  int mtile = w >> 4;                  // 256 m-tiles of 32 over B*SQ
  int ntile = w & 15;                  // 16 n-tiles of 64 over D
  int b  = mtile >> 6;                 // SQ_/32 = 64 m-tiles per batch
  int m0 = mtile * 32;
  int n0 = ntile * 64;
  int lane = threadIdx.x & 31;

  const __bf16* a0p = P + (size_t)m0 * SK_;
  const __bf16* a1p = P + (size_t)(m0 + 16) * SK_;
  const __bf16* bp  = Vt + (size_t)b * D_ * SK_ + (size_t)n0 * SK_;

  v8f acc[2][4] = {};

  v16bf a0 = load_a_bf16(a0p, SK_);
  v16bf a1 = load_a_bf16(a1p, SK_);
  v16bf bfr[4];
#pragma unroll
  for (int j = 0; j < 4; ++j) bfr[j] = load_b_bf16(bp + (size_t)j * 16 * SK_, SK_);

  for (int kk = 0; kk < SK_ - 32; kk += 32) {
    int nk = kk + 32;
    __builtin_prefetch(a0p + (size_t)(lane & 15) * SK_ + nk + 32, 0, 3);
    v16bf na0 = load_a_bf16(a0p + nk, SK_);
    v16bf na1 = load_a_bf16(a1p + nk, SK_);
    v16bf nb[4];
#pragma unroll
    for (int j = 0; j < 4; ++j) nb[j] = load_b_bf16(bp + (size_t)j * 16 * SK_ + nk, SK_);
#pragma unroll
    for (int j = 0; j < 4; ++j) {
      acc[0][j] = wmma_bf16(a0, bfr[j], acc[0][j]);
      acc[1][j] = wmma_bf16(a1, bfr[j], acc[1][j]);
    }
    a0 = na0; a1 = na1;
#pragma unroll
    for (int j = 0; j < 4; ++j) bfr[j] = nb[j];
  }
#pragma unroll
  for (int j = 0; j < 4; ++j) {
    acc[0][j] = wmma_bf16(a0, bfr[j], acc[0][j]);
    acc[1][j] = wmma_bf16(a1, bfr[j], acc[1][j]);
  }

  int rofs = (lane < 16) ? 0 : 8;
  int c    = lane & 15;
#pragma unroll
  for (int i = 0; i < 2; ++i)
#pragma unroll
    for (int j = 0; j < 4; ++j)
#pragma unroll
      for (int r = 0; r < 8; ++r)
        out[(size_t)(m0 + i * 16 + r + rofs) * D_ + n0 + j * 16 + c] = acc[i][j][r];
}

// ---------------------------------------------------------------------------
extern "C" void kernel_launch(void* const* d_in, const int* in_sizes, int n_in,
                              void* d_out, int out_size, void* d_ws, size_t ws_size,
                              hipStream_t stream) {
  const float* q    = (const float*)d_in[0];
  const float* k    = (const float*)d_in[1];
  const float* v    = (const float*)d_in[2];
  const float* W1_w = (const float*)d_in[3];
  const float* W1_b = (const float*)d_in[4];
  const float* W2_w = (const float*)d_in[5];
  const float* W2_b = (const float*)d_in[6];
  float* out = (float*)d_out;

  // Workspace layout (all bf16 intermediates; ~84 MB total, L2-resident):
  //   [0,2M)    W1t [U][D]       [2M,4M)   W2t [U][D]
  //   [4M,20M)  l1  [B*SQ][U]    [20M,36M) l2  [B*SK][U]
  //   [36M,52M) Vt  [B][D][SK]   [52M,84M) P   [B*SQ][SK]
  char* ws = (char*)d_ws;
  __bf16* w1t = (__bf16*)(ws);
  __bf16* w2t = (__bf16*)(ws + (size_t)2  * 1024 * 1024);
  __bf16* l1  = (__bf16*)(ws + (size_t)4  * 1024 * 1024);
  __bf16* l2  = (__bf16*)(ws + (size_t)20 * 1024 * 1024);
  __bf16* vt  = (__bf16*)(ws + (size_t)36 * 1024 * 1024);
  __bf16* Pm  = (__bf16*)(ws + (size_t)52 * 1024 * 1024);

  // Allow the fused score+softmax kernel its ~161KB of dynamic LDS
  // (CDNA5 WGP supports 320KB per workgroup).  Deterministic, capture-safe.
  size_t shmem = (size_t)16 * U_ * 2 + (size_t)16 * SK_ * 4 + 16 * 16 * 4;
  (void)hipFuncSetAttribute((const void*)k_score_softmax,
                            hipFuncAttributeMaxDynamicSharedMemorySize, (int)shmem);

  k_transpose_w<<<(U_ * D_) / 256, 256, 0, stream>>>(W1_w, w1t);
  k_transpose_w<<<(U_ * D_) / 256, 256, 0, stream>>>(W2_w, w2t);
  k_transpose_v<<<(B_ * D_ * SK_) / 256, 256, 0, stream>>>(v, vt);

  // 4096 waves each: 256 m-tiles x 16 n-tiles; 8 waves per 256-thread block.
  k_proj<<<512, 256, 0, stream>>>(q, w1t, W1_b, l1);
  k_proj<<<512, 256, 0, stream>>>(k, w2t, W2_b, l2);

  k_score_softmax<<<B_ * (SQ_ / 16), 256, shmem, stream>>>(l1, l2, Pm);

  k_att<<<512, 256, 0, stream>>>(Pm, vt, out);
}
